// TransformerBlock_56624848830998
// MI455X (gfx1250) — matrix operand order repaired and checked by reading, not compiled, not run on hardware
//
#include <hip/hip_runtime.h>
#include <hip/hip_bf16.h>

// ---------------------------------------------------------------------------
// Transformer block for MI455X (gfx1250, wave32, WMMA).
// Pipeline:
//   1. k_cvt        : fp32 weights -> bf16 (Wq,Wk,Wv,Wo,W1,W2)
//   2. k_ln         : LN1(inp) -> x_f32 (residual) + x_bf16 (GEMM A)
//   3. k_gemm<0> x3 : Q/K/V = x@W + b  -> bf16 [B,H,S,D]
//   4. k_attn       : flash attention  -> attn bf16 [B,S,E]
//   5. k_gemm<1>    : y = attn@Wo + bo + x_f32           (fp32)
//   6. k_ln         : LN2(y) in-place  -> y_f32 + h_bf16
//   7. k_gemm<2>    : h1 = gelu(h@W1 + b1)               (bf16)
//   8. k_gemm<1>    : out = h1@W2 + b2 + y_f32           -> d_out fp32
//
// GEMM: 128x128 block tile, 8 waves (32x64 each -> 8 WMMA/K-step/wave),
// double-buffered LDS fed by GLOBAL_LOAD_ASYNC_TO_LDS_B128 (ASYNCcnt-fenced),
// sync-copy fallback if the builtin is absent.
// ---------------------------------------------------------------------------

typedef __attribute__((ext_vector_type(16))) __bf16 v16bf;
typedef __attribute__((ext_vector_type(8)))  __bf16 v8bf;
typedef __attribute__((ext_vector_type(4)))  __bf16 v4bf;
typedef __attribute__((ext_vector_type(8)))  float  v8f;
typedef int v4i_gcc __attribute__((vector_size(16)));

#if __has_builtin(__builtin_amdgcn_global_load_async_to_lds_b128) && \
    __has_builtin(__builtin_amdgcn_s_wait_asynccnt)
#define ASYNC_LDS 1
#else
#define ASYNC_LDS 0
#endif

// 16-byte global -> LDS copy (async DMA when available)
__device__ __forceinline__ void cp16_g2l(__bf16* dst, const __bf16* src) {
#if ASYNC_LDS
    __builtin_amdgcn_global_load_async_to_lds_b128(
        (__attribute__((address_space(1))) v4i_gcc*)(const_cast<__bf16*>(src)),
        (__attribute__((address_space(3))) v4i_gcc*)dst, 0, 0);
#else
    *(v8bf*)dst = *(const v8bf*)src;
#endif
}
__device__ __forceinline__ void async_fence() {
#if ASYNC_LDS
    __builtin_amdgcn_s_wait_asynccnt(0);
#endif
}

__device__ __forceinline__ v16bf cat8(v8bf lo, v8bf hi) {
    v16bf r;
#pragma unroll
    for (int i = 0; i < 8; ++i) { r[i] = lo[i]; r[i + 8] = hi[i]; }
    return r;
}

// ------------------------------ fp32 -> bf16 -------------------------------
__global__ void k_cvt(const float* __restrict__ in, __bf16* __restrict__ out, int n) {
    int i = blockIdx.x * blockDim.x + threadIdx.x;
    if (i * 4 < n) {
        float4 v = ((const float4*)in)[i];
        v4bf o;
        o[0] = (__bf16)v.x; o[1] = (__bf16)v.y; o[2] = (__bf16)v.z; o[3] = (__bf16)v.w;
        ((v4bf*)out)[i] = o;
    }
}

// ------------------------------ LayerNorm (E=1024) -------------------------
__global__ __launch_bounds__(256)
void k_ln(const float* __restrict__ in, const float* __restrict__ g,
          const float* __restrict__ bta, float* __restrict__ outf,
          __bf16* __restrict__ outb) {
    const int row = blockIdx.x, tid = threadIdx.x;
    const float* x = in + (size_t)row * 1024;
    float4 v = ((const float4*)x)[tid];
    float s  = v.x + v.y + v.z + v.w;
    float s2 = v.x * v.x + v.y * v.y + v.z * v.z + v.w * v.w;
#pragma unroll
    for (int off = 16; off >= 1; off >>= 1) {
        s  += __shfl_xor(s,  off, 32);
        s2 += __shfl_xor(s2, off, 32);
    }
    __shared__ float red[16];
    __shared__ float stats[2];
    const int wv = tid >> 5;
    if ((tid & 31) == 0) { red[wv] = s; red[wv + 8] = s2; }
    __syncthreads();
    if (tid == 0) {
        float ts = 0.f, ts2 = 0.f;
        for (int i = 0; i < 8; ++i) { ts += red[i]; ts2 += red[i + 8]; }
        float mu  = ts * (1.0f / 1024.0f);
        float var = ts2 * (1.0f / 1024.0f) - mu * mu;
        stats[0] = mu;
        stats[1] = rsqrtf(var + 1e-5f);
    }
    __syncthreads();
    float mu = stats[0], rs = stats[1];
    float4 gv = ((const float4*)g)[tid];
    float4 bv = ((const float4*)bta)[tid];
    float4 y;
    y.x = (v.x - mu) * rs * gv.x + bv.x;
    y.y = (v.y - mu) * rs * gv.y + bv.y;
    y.z = (v.z - mu) * rs * gv.z + bv.z;
    y.w = (v.w - mu) * rs * gv.w + bv.w;
    ((float4*)(outf + (size_t)row * 1024))[tid] = y;
    v4bf ob;
    ob[0] = (__bf16)y.x; ob[1] = (__bf16)y.y; ob[2] = (__bf16)y.z; ob[3] = (__bf16)y.w;
    *(v4bf*)(outb + (size_t)row * 1024 + tid * 4) = ob;
}

// ------------------------------ WMMA GEMM ----------------------------------
// C[M,N] = A[M,K]@B[K,N] + bias, A/B bf16 row-major.
// Block: 256 thr (8 waves), 128x128 tile; wave = 32x64 (2x4 WMMA); BK=32.
// Double-buffered LDS, async global->LDS A-tile copies.
// MODE 0: bf16 store remapped to [B,H,S,D]  (QKV)
// MODE 1: fp32 store + residual             (Wo proj, MLP2)
// MODE 2: bf16 store with exact GELU        (MLP1)
template <int MODE>
__global__ __launch_bounds__(256)
void k_gemm(const __bf16* __restrict__ A, const __bf16* __restrict__ Bm,
            const float* __restrict__ bias, int M, int N, int K,
            float* __restrict__ outf, __bf16* __restrict__ outb,
            const float* __restrict__ res) {
    __shared__ __bf16 sA[2][128 * 40];   // [row][k], pad 40 (80B rows, 16B aligned)
    __shared__ __bf16 sBt[2][128 * 40];  // [n][k] transposed

    const int tid = threadIdx.x;
    const int wave = tid >> 5, lane = tid & 31;
    const int half = lane >> 4, l16 = lane & 15;
    const int bm = blockIdx.y * 128, bn = blockIdx.x * 128;
    const int wm = (wave >> 1) * 32, wn = (wave & 1) * 64;

    v8f acc[2][4] = {};

    const int arow = tid >> 1, aseg = (tid & 1) * 16;   // A: 128 rows x 32 k
    const int btn = tid & 127, btk = (tid >> 7) * 16;   // B: 32 k x 128 n

    const __bf16* gA = A + (size_t)(bm + arow) * K + aseg;

    // prologue: stage tile 0 into buffer 0
    cp16_g2l(&sA[0][arow * 40 + aseg], gA);
    cp16_g2l(&sA[0][arow * 40 + aseg + 8], gA + 8);
#pragma unroll
    for (int i = 0; i < 16; ++i)
        sBt[0][btn * 40 + btk + i] = Bm[(size_t)(btk + i) * N + bn + btn];
    async_fence();
    __syncthreads();

    const int ksteps = K >> 5;
    for (int kt = 0; kt < ksteps; ++kt) {
        const int cur = kt & 1, nxt = cur ^ 1;
        const bool hasNext = (kt + 1) < ksteps;
        __bf16 breg[16];
        if (hasNext) {
            const int k0n = (kt + 1) << 5;
            cp16_g2l(&sA[nxt][arow * 40 + aseg], gA + k0n);
            cp16_g2l(&sA[nxt][arow * 40 + aseg + 8], gA + k0n + 8);
#pragma unroll
            for (int i = 0; i < 16; ++i)
                breg[i] = Bm[(size_t)(k0n + btk + i) * N + bn + btn];
        }

        // compute on current buffer
        v16bf af[2], bf4[4];
#pragma unroll
        for (int i = 0; i < 2; ++i) {
            const __bf16* p = &sA[cur][(wm + i * 16 + l16) * 40];
            af[i] = cat8(*(const v8bf*)(p + half * 8),
                         *(const v8bf*)(p + 16 + half * 8));
        }
#pragma unroll
        for (int j = 0; j < 4; ++j) {
            const __bf16* p = &sBt[cur][(wn + j * 16 + l16) * 40 + half * 8];
            bf4[j] = cat8(*(const v8bf*)p, *(const v8bf*)(p + 16));
        }
#pragma unroll
        for (int i = 0; i < 2; ++i)
#pragma unroll
            for (int j = 0; j < 4; ++j)
                acc[i][j] = __builtin_amdgcn_wmma_f32_16x16x32_bf16(
                    false, af[i], false, bf4[j], (short)0, acc[i][j], false, false);

        if (hasNext) {
#pragma unroll
            for (int i = 0; i < 16; ++i)
                sBt[nxt][btn * 40 + btk + i] = breg[i];
            async_fence();
            __syncthreads();
        }
    }

    // epilogue: row m = base + half*8 + r, col n = base + l16 (ISA C/D layout)
#pragma unroll
    for (int i = 0; i < 2; ++i)
#pragma unroll
        for (int j = 0; j < 4; ++j) {
            int n = bn + wn + j * 16 + l16;
            float bv = bias[n];
#pragma unroll
            for (int r = 0; r < 8; ++r) {
                int m = bm + wm + i * 16 + half * 8 + r;
                float c = acc[i][j][r] + bv;
                if constexpr (MODE == 0) {
                    int bb = m >> 11, s = m & 2047, h = n >> 6, d = n & 63;
                    outb[((((size_t)bb * 16 + h) << 11) + s) * 64 + d] = (__bf16)c;
                } else if constexpr (MODE == 1) {
                    size_t idx = (size_t)m * N + n;
                    outf[idx] = c + res[idx];
                } else {
                    float gx = 0.5f * c * (1.0f + erff(c * 0.70710678f));
                    outb[(size_t)m * N + n] = (__bf16)gx;
                }
            }
        }
}

// ------------------------------ Flash attention ----------------------------
// grid(32, 64): x = q-tile (64 rows), y = b*16+h. 128 thr, wave = 16 q-rows.
__global__ __launch_bounds__(128)
void k_attn(const __bf16* __restrict__ Q, const __bf16* __restrict__ Kp,
            const __bf16* __restrict__ V, __bf16* __restrict__ out) {
    __shared__ __bf16 sVt[64 * 72];       // V tile transposed [d][key]
    __shared__ __bf16 sP[4 * 16 * 72];    // per-wave P tile (C-layout -> A-layout)

    const int tid = threadIdx.x, wave = tid >> 5, lane = tid & 31;
    const int half = lane >> 4, l16 = lane & 15;
    const int bh = blockIdx.y;
    const int qbase = blockIdx.x * 64 + wave * 16;
    const __bf16* Qb = Q  + (size_t)bh * 2048 * 64;
    const __bf16* Kb = Kp + (size_t)bh * 2048 * 64;
    const __bf16* Vb = V  + (size_t)bh * 2048 * 64;

    // Q fragments (16 rows x 64 d = two 16x32 A-frags), straight from global
    v16bf qf[2];
    {
        const __bf16* qp = Qb + (size_t)(qbase + l16) * 64;
#pragma unroll
        for (int ks = 0; ks < 2; ++ks)
            qf[ks] = cat8(*(const v8bf*)(qp + ks * 32 + half * 8),
                          *(const v8bf*)(qp + ks * 32 + 16 + half * 8));
    }

    v8f o[4] = {};
    float mrow[8], lrow[8];
#pragma unroll
    for (int j = 0; j < 8; ++j) { mrow[j] = -1e30f; lrow[j] = 0.f; }

    __bf16* pb = &sP[wave * 16 * 72];
    const int vd = tid & 63, vkh = (tid >> 6) * 32;

    for (int kb = 0; kb < 2048; kb += 64) {
        // V tile transposed into LDS (coalesced global read across d)
#pragma unroll 4
        for (int i = 0; i < 32; ++i)
            sVt[vd * 72 + vkh + i] = Vb[(size_t)(kb + vkh + i) * 64 + vd];
        __syncthreads();

        // scores: 16x64 tile = 4 WMMA cols, K-dim = 64 (2 steps).
        // B-frag of K^T == row-major K[key][d] -> direct global gather.
        v8f sc[4] = {};
#pragma unroll
        for (int nt = 0; nt < 4; ++nt)
#pragma unroll
            for (int ks = 0; ks < 2; ++ks) {
                const __bf16* kp =
                    Kb + (size_t)(kb + nt * 16 + l16) * 64 + ks * 32 + half * 8;
                v16bf bfr = cat8(*(const v8bf*)kp, *(const v8bf*)(kp + 16));
                sc[nt] = __builtin_amdgcn_wmma_f32_16x16x32_bf16(
                    false, qf[ks], false, bfr, (short)0, sc[nt], false, false);
            }

        // online softmax: row j lives in acc element j, lanes of one 16-half
#pragma unroll
        for (int j = 0; j < 8; ++j) {
            float v = -1e30f;
#pragma unroll
            for (int nt = 0; nt < 4; ++nt) {
                sc[nt][j] *= 0.125f;  // 1/sqrt(64)
                v = fmaxf(v, sc[nt][j]);
            }
#pragma unroll
            for (int off = 8; off >= 1; off >>= 1)
                v = fmaxf(v, __shfl_xor(v, off, 32));
            float mnew  = fmaxf(mrow[j], v);
            float alpha = __expf(mrow[j] - mnew);
            float s = 0.f;
#pragma unroll
            for (int nt = 0; nt < 4; ++nt) {
                float p = __expf(sc[nt][j] - mnew);
                sc[nt][j] = p;
                s += p;
            }
#pragma unroll
            for (int off = 8; off >= 1; off >>= 1) s += __shfl_xor(s, off, 32);
            lrow[j] = lrow[j] * alpha + s;
            mrow[j] = mnew;
#pragma unroll
            for (int nt = 0; nt < 4; ++nt) o[nt][j] *= alpha;
            // P C-layout -> LDS row-major (row = half*8+j)
#pragma unroll
            for (int nt = 0; nt < 4; ++nt)
                pb[(half * 8 + j) * 72 + nt * 16 + l16] = (__bf16)sc[nt][j];
        }

        // O += P @ V  (A-frags from pb, B-frags from transposed V tile)
#pragma unroll
        for (int ks = 0; ks < 2; ++ks) {
            const __bf16* ap = pb + l16 * 72 + ks * 32;
            v16bf pf = cat8(*(const v8bf*)(ap + half * 8),
                            *(const v8bf*)(ap + 16 + half * 8));
#pragma unroll
            for (int nt = 0; nt < 4; ++nt) {
                const __bf16* vp =
                    &sVt[(nt * 16 + l16) * 72 + ks * 32 + half * 8];
                v16bf vf = cat8(*(const v8bf*)vp, *(const v8bf*)(vp + 16));
                o[nt] = __builtin_amdgcn_wmma_f32_16x16x32_bf16(
                    false, pf, false, vf, (short)0, o[nt], false, false);
            }
        }
        __syncthreads();
    }

    // normalize + store to [B,S,E] bf16 (feeds Wo GEMM as A)
    const int b = bh >> 4, h = bh & 15;
#pragma unroll
    for (int j = 0; j < 8; ++j) {
        float inv = 1.0f / lrow[j];
        int s = qbase + half * 8 + j;
#pragma unroll
        for (int nt = 0; nt < 4; ++nt) {
            int e = h * 64 + nt * 16 + l16;
            out[((size_t)b * 2048 + s) * 1024 + e] = (__bf16)(o[nt][j] * inv);
        }
    }
}

// ------------------------------ launcher -----------------------------------
extern "C" void kernel_launch(void* const* d_in, const int* in_sizes, int n_in,
                              void* d_out, int out_size, void* d_ws, size_t ws_size,
                              hipStream_t stream) {
    (void)in_sizes; (void)n_in; (void)out_size; (void)ws_size;
    const float* inp  = (const float*)d_in[0];
    const float* ln1g = (const float*)d_in[1];
    const float* ln1b = (const float*)d_in[2];
    const float* Wq = (const float*)d_in[3];   const float* bq = (const float*)d_in[4];
    const float* Wk = (const float*)d_in[5];   const float* bk = (const float*)d_in[6];
    const float* Wv = (const float*)d_in[7];   const float* bv = (const float*)d_in[8];
    const float* Wo = (const float*)d_in[9];   const float* bo = (const float*)d_in[10];
    const float* ln2g = (const float*)d_in[11];
    const float* ln2b = (const float*)d_in[12];
    const float* W1 = (const float*)d_in[13];  const float* b1 = (const float*)d_in[14];
    const float* W2 = (const float*)d_in[15];  const float* b2 = (const float*)d_in[16];

    const size_t M = 8192, E = 1024, E2 = 2048;
    char* ws = (char*)d_ws;
    size_t off = 0;
    auto take = [&](size_t bytes) {
        char* p = ws + off;
        off += (bytes + 255) & ~(size_t)255;
        return p;
    };
    float*  x_f32  = (float*) take(M * E * 4);
    __bf16* x_bf   = (__bf16*)take(M * E * 2);
    __bf16* wq_bf  = (__bf16*)take(E * E * 2);
    __bf16* wk_bf  = (__bf16*)take(E * E * 2);
    __bf16* wv_bf  = (__bf16*)take(E * E * 2);
    __bf16* wo_bf  = (__bf16*)take(E * E * 2);
    __bf16* w1_bf  = (__bf16*)take(E * E2 * 2);
    __bf16* w2_bf  = (__bf16*)take(E2 * E * 2);
    __bf16* Qh     = (__bf16*)take(M * E * 2);
    __bf16* Kh     = (__bf16*)take(M * E * 2);
    __bf16* Vh     = (__bf16*)take(M * E * 2);
    __bf16* att_bf = (__bf16*)take(M * E * 2);
    float*  y_f32  = (float*) take(M * E * 4);
    __bf16* h_bf   = (__bf16*)take(M * E * 2);
    __bf16* h1_bf  = (__bf16*)take(M * E2 * 2);

    // 1. weight downconverts
    int cvtBlocks = (int)((E * E / 4 + 255) / 256);
    k_cvt<<<cvtBlocks, 256, 0, stream>>>(Wq, wq_bf, (int)(E * E));
    k_cvt<<<cvtBlocks, 256, 0, stream>>>(Wk, wk_bf, (int)(E * E));
    k_cvt<<<cvtBlocks, 256, 0, stream>>>(Wv, wv_bf, (int)(E * E));
    k_cvt<<<cvtBlocks, 256, 0, stream>>>(Wo, wo_bf, (int)(E * E));
    k_cvt<<<cvtBlocks * 2, 256, 0, stream>>>(W1, w1_bf, (int)(E * E2));
    k_cvt<<<cvtBlocks * 2, 256, 0, stream>>>(W2, w2_bf, (int)(E2 * E));

    // 2. LN1
    k_ln<<<(int)M, 256, 0, stream>>>(inp, ln1g, ln1b, x_f32, x_bf);

    // 3. QKV projections -> [B,H,S,D]
    dim3 gE(E / 128, M / 128);
    k_gemm<0><<<gE, 256, 0, stream>>>(x_bf, wq_bf, bq, (int)M, (int)E, (int)E,
                                      nullptr, Qh, nullptr);
    k_gemm<0><<<gE, 256, 0, stream>>>(x_bf, wk_bf, bk, (int)M, (int)E, (int)E,
                                      nullptr, Kh, nullptr);
    k_gemm<0><<<gE, 256, 0, stream>>>(x_bf, wv_bf, bv, (int)M, (int)E, (int)E,
                                      nullptr, Vh, nullptr);

    // 4. flash attention
    k_attn<<<dim3(32, 64), 128, 0, stream>>>(Qh, Kh, Vh, att_bf);

    // 5. output projection + residual (fp32)
    k_gemm<1><<<gE, 256, 0, stream>>>(att_bf, wo_bf, bo, (int)M, (int)E, (int)E,
                                      y_f32, nullptr, x_f32);

    // 6. LN2 (in-place fp32) + bf16 for MLP
    k_ln<<<(int)M, 256, 0, stream>>>(y_f32, ln2g, ln2b, y_f32, h_bf);

    // 7. MLP up + GELU
    dim3 gE2(E2 / 128, M / 128);
    k_gemm<2><<<gE2, 256, 0, stream>>>(h_bf, w1_bf, b1, (int)M, (int)E2, (int)E,
                                       nullptr, h1_bf, nullptr);

    // 8. MLP down + residual -> d_out
    k_gemm<1><<<gE, 256, 0, stream>>>(h1_bf, w2_bf, b2, (int)M, (int)E, (int)E2,
                                      (float*)d_out, nullptr, y_f32);
}
